// RWKV6Block_65816078844131
// MI455X (gfx1250) — matrix operand order repaired and checked by reading, not compiled, not run on hardware
//
#include <hip/hip_runtime.h>

// ---------------------------------------------------------------------------
// RWKV6 block for MI455X (gfx1250, wave32). All dense projections run through
// v_wmma_f32_16x16x32_bf16 (bf16 in, fp32 accumulate). GEMM tiles are fed by
// the Tensor Data Mover (tensor_load_to_lds, TENSORcnt) with double-buffered
// LDS. The sequential WKV recurrence stays in fp32 VALU. Nonlinearities are
// fused into GEMM epilogues.
// ---------------------------------------------------------------------------

typedef __bf16 bf16;
typedef __attribute__((ext_vector_type(16))) __bf16 v16bf;
typedef __attribute__((ext_vector_type(8)))  float  v8f;
typedef __attribute__((ext_vector_type(4)))  unsigned int v4u;
typedef __attribute__((ext_vector_type(8)))  int  v8i;
typedef __attribute__((ext_vector_type(4)))  int  v4i;

#define T_SEQ 2048
#define H_DIM 1024
#define NH_   16
#define D_    64
#define FF_   4096

__device__ __forceinline__ bf16 tobf(float f) { return (bf16)f; }

// ---------------------------------------------------------------------------
// TDM: DMA a 2D tile (tile_d1 rows x tile_d0 elements, 2-byte elements) from
// global memory into LDS. Descriptor per CDNA5 ISA sec 8 (D# groups 0/1;
// groups 2/3 zero for <=2D tensors). tensor_d0/d1 are the remaining extents
// from the tile origin so OOB rows/cols read as zero (ragged-N handling).
// This toolchain exposes the 6-arg builtin (g0, g1, g2, g3, g_extra, cpol).
// ---------------------------------------------------------------------------
__device__ __forceinline__ void tdm_load_2d(
    unsigned lds_addr, const void* gaddr,
    unsigned tensor_d0, unsigned tensor_d1,
    unsigned tile_d0, unsigned tile_d1,
    unsigned stride0 /* elements */)
{
  const unsigned long long ga = (unsigned long long)gaddr;
  v4u g0;
  g0[0] = 1u;                                            // count=1 (valid user D#)
  g0[1] = lds_addr;                                      // lds_addr[31:0]
  g0[2] = (unsigned)(ga & 0xffffffffu);                  // global_addr[31:0]
  g0[3] = (unsigned)((ga >> 32) & 0x01ffffffu)           // global_addr[56:32]
          | (2u << 30);                                  // type = 2 ("image")
  v8i g1;
  g1[0] = (int)(1u << 16);                               // data_size=1 (2 bytes)
  g1[1] = (int)((tensor_d0 & 0xffffu) << 16);            // tensor_dim0[15:0]
  g1[2] = (int)((tensor_d0 >> 16) |
                ((tensor_d1 & 0xffffu) << 16));          // td0[31:16] | td1[15:0]
  g1[3] = (int)((tensor_d1 >> 16) | (tile_d0 << 16));    // td1[31:16] | tile_dim0
  g1[4] = (int)tile_d1;                                  // tile_dim1 (tile_dim2=0)
  g1[5] = (int)stride0;                                  // tensor_dim0_stride[31:0]
  g1[6] = 0;                                             // stride0[47:32]|stride1 lo
  g1[7] = 0;
  const v4i gz4 = {0, 0, 0, 0};
  const v8i gz8 = {0, 0, 0, 0, 0, 0, 0, 0};
  __builtin_amdgcn_tensor_load_to_lds(g0, g1, gz4, gz4, gz8, 0);
}

enum {
  E_NONE = 0,   // fp32 out = acc
  E_TANH_BF,    // bf16 out = tanh(acc)
  E_MIX,        // bf16 out = h + delta*(maa + acc)           (p0=h, p1=delta, p2=maa)
  E_SILU,       // fp32 out = acc * sigmoid(acc)
  E_SIGSCALE,   // fp32 out = sigmoid(acc) * D^-0.5
  E_DECAY,      // fp32 out = exp(-exp(time_decay[col]+acc))  (p0=time_decay)
  E_ADDRES,     // fp32 out = res + acc                        (p0=res)
  E_RELU2_BF,   // bf16 out = relu(acc)^2
  E_SIGMOID,    // fp32 out = sigmoid(acc)
  E_FFNOUT      // fp32 out = res + sig*acc                    (p0=res, p1=sig)
};

// ---------------------------------------------------------------------------
// Generic bf16 GEMM: C[M,N] = A[M,K] (row-major, stride lda) x Bt[N,K]
// (B pre-transposed so its WMMA fragment is one contiguous 32B LDS load).
// Block tile 64x64, 4 waves (128 threads), BK = 32. Tile feed = TDM DMA with
// double-buffered LDS: wave 0 issues next-tile DMAs, all waves compute the
// current buffer, then s_wait_tensorcnt + barrier rotates the buffers.
// M is always a multiple of 64 here; N may be ragged (TDM zero-fills OOB).
// ---------------------------------------------------------------------------
template<int EPI>
__global__ __launch_bounds__(128) void gemm_bf16_kernel(
    const bf16* __restrict__ A, int lda,
    const bf16* __restrict__ Bt,
    int M, int N, int K,
    float* __restrict__ outF, bf16* __restrict__ outB, int ldc,
    const float* __restrict__ p0, const float* __restrict__ p1,
    const float* __restrict__ p2)
{
  __shared__ alignas(64) bf16 As[2][64][32];
  __shared__ alignas(64) bf16 Bs[2][64][32];

  const int tid  = threadIdx.x;
  const int wave = tid >> 5;
  const int lane = tid & 31;
  const int kh   = lane >> 4;   // lane-half: selects K sub-group (ISA 7.12.2)
  const int ln   = lane & 15;
  const int m0   = blockIdx.x * 64;
  const int n0   = blockIdx.y * 64;

  // generic->LDS offset: low 32 bits of the flat address are the LDS address
  const unsigned ldsA = (unsigned)(unsigned long long)(const void*)&As[0][0][0];
  const unsigned ldsB = (unsigned)(unsigned long long)(const void*)&Bs[0][0][0];
  const unsigned rowsA = (unsigned)(M - m0);
  const unsigned rowsB = (N > n0) ? (unsigned)(N - n0) : 0u;

  v8f acc[4] = {};

  // prologue: DMA first K-tile into buffer 0
  if (tid < 32) {
    tdm_load_2d(ldsA, A + (size_t)m0 * lda, (unsigned)K, rowsA, 32, 64, (unsigned)lda);
    tdm_load_2d(ldsB, Bt + (size_t)n0 * K, (unsigned)K, rowsB, 32, 64, (unsigned)K);
    __builtin_amdgcn_s_wait_tensorcnt(0);
  }
  __syncthreads();

  int buf = 0;
  for (int k0 = 0; k0 < K; k0 += 32) {
    const int k1 = k0 + 32;
    if (k1 < K && tid < 32) {  // kick off DMA of the next tile into buf^1
      tdm_load_2d(ldsA + (unsigned)((buf ^ 1) * 4096),
                  A + (size_t)m0 * lda + k1, (unsigned)(K - k1), rowsA,
                  32, 64, (unsigned)lda);
      tdm_load_2d(ldsB + (unsigned)((buf ^ 1) * 4096),
                  Bt + (size_t)n0 * K + k1, (unsigned)(K - k1), rowsB,
                  32, 64, (unsigned)K);
    }

    // A fragment 16x32: lanes 0-15 / 16-31 both cover M=0..15; K grouping:
    // element i -> K = kh*16 + i (contiguous, matches B).
    v16bf af = *(const v16bf*)&As[buf][wave * 16 + ln][kh * 16];
    // B fragment 32x16: N = lane&15, element i -> K = kh*16 + i (contiguous)
    #pragma unroll
    for (int nt = 0; nt < 4; nt++) {
      v16bf bfrag = *(const v16bf*)&Bs[buf][nt * 16 + ln][kh * 16];
      acc[nt] = __builtin_amdgcn_wmma_f32_16x16x32_bf16(
          false, af, false, bfrag, (short)0, acc[nt], false, false);
    }

    if (tid < 32) __builtin_amdgcn_s_wait_tensorcnt(0);
    __syncthreads();
    buf ^= 1;
  }

  // C/D layout: vgpr e -> M = e + kh*8 ; N = lane&15
  const int rowBase = m0 + wave * 16 + kh * 8;
  #pragma unroll
  for (int nt = 0; nt < 4; nt++) {
    const int col = n0 + nt * 16 + ln;
    if (col >= N) continue;
    #pragma unroll
    for (int e = 0; e < 8; e++) {
      const int row = rowBase + e;
      const float a = acc[nt][e];
      const size_t oi = (size_t)row * ldc + col;
      if (EPI == E_NONE) {
        outF[oi] = a;
      } else if (EPI == E_TANH_BF) {
        outB[oi] = tobf(tanhf(a));
      } else if (EPI == E_MIX) {
        const size_t hi = (size_t)row * H_DIM + col;
        outB[oi] = tobf(p0[hi] + p1[hi] * (p2[col] + a));
      } else if (EPI == E_SILU) {
        outF[oi] = a / (1.0f + __expf(-a));
      } else if (EPI == E_SIGSCALE) {
        outF[oi] = 0.125f / (1.0f + __expf(-a));   // sigmoid(a) * D^-0.5
      } else if (EPI == E_DECAY) {
        outF[oi] = __expf(-__expf(p0[col] + a));
      } else if (EPI == E_ADDRES) {
        outF[oi] = p0[oi] + a;
      } else if (EPI == E_RELU2_BF) {
        const float r = a > 0.0f ? a : 0.0f;
        outB[oi] = tobf(r * r);
      } else if (EPI == E_SIGMOID) {
        outF[oi] = 1.0f / (1.0f + __expf(-a));
      } else if (EPI == E_FFNOUT) {
        const size_t hi = (size_t)row * H_DIM + col;
        outF[oi] = p0[hi] + p1[hi] * a;
      }
    }
  }
}

// fp32 weight [K][N] -> bf16 transposed [N][K]
__global__ void wconv_kernel(const float* __restrict__ W, bf16* __restrict__ Wt,
                             int K, int N)
{
  const int idx = blockIdx.x * blockDim.x + threadIdx.x;
  if (idx >= K * N) return;
  const int k = idx / N, n = idx % N;
  Wt[(size_t)n * K + k] = tobf(W[idx]);
}

// RMSNorm per row (block per token)
__global__ __launch_bounds__(256) void rmsnorm_kernel(
    const float* __restrict__ x, const float* __restrict__ w,
    float* __restrict__ out)
{
  const int t = blockIdx.x;
  __shared__ float red[256];
  float s = 0.0f;
  for (int c = threadIdx.x; c < H_DIM; c += 256) {
    const float v = x[(size_t)t * H_DIM + c];
    s += v * v;
  }
  red[threadIdx.x] = s;
  __syncthreads();
  for (int o = 128; o > 0; o >>= 1) {
    if (threadIdx.x < o) red[threadIdx.x] += red[threadIdx.x + o];
    __syncthreads();
  }
  const float inv = rsqrtf(red[0] * (1.0f / H_DIM) + 1e-6f);
  for (int c = threadIdx.x; c < H_DIM; c += 256)
    out[(size_t)t * H_DIM + c] = x[(size_t)t * H_DIM + c] * inv * w[c];
}

// token shift stage 1: delta = h[t-1]-h[t]; xxx = h + delta*maa_x (bf16)
__global__ void shift1_kernel(const float* __restrict__ h,
                              const float* __restrict__ maa_x,
                              float* __restrict__ delta,
                              bf16* __restrict__ xxx, int total)
{
  const int idx = blockIdx.x * blockDim.x + threadIdx.x;
  if (idx >= total) return;
  const int t = idx / H_DIM, c = idx % H_DIM;
  const float cur  = h[idx];
  const float prev = (t == 0) ? 0.0f : h[idx - H_DIM];
  const float d = prev - cur;
  delta[idx] = d;
  xxx[idx]   = tobf(cur + d * maa_x[c]);
}

// token shift stage 2 (FFN): xk2/xr2 in bf16
__global__ void shift2_kernel(const float* __restrict__ h,
                              const float* __restrict__ maa_k,
                              const float* __restrict__ maa_r,
                              bf16* __restrict__ xk2,
                              bf16* __restrict__ xr2, int total)
{
  const int idx = blockIdx.x * blockDim.x + threadIdx.x;
  if (idx >= total) return;
  const int t = idx / H_DIM, c = idx % H_DIM;
  const float cur  = h[idx];
  const float prev = (t == 0) ? 0.0f : h[idx - H_DIM];
  const float d = prev - cur;
  xk2[idx] = tobf(cur + d * maa_k[c]);
  xr2[idx] = tobf(cur + d * maa_r[c]);
}

// Sequential WKV scan: one block per head, one thread per v-column,
// 64-entry fp32 state kept in registers, r/k/decay broadcast via LDS.
__global__ __launch_bounds__(64) void scan_kernel(
    const float* __restrict__ r, const float* __restrict__ k,
    const float* __restrict__ v, const float* __restrict__ ew,
    const float* __restrict__ u, float* __restrict__ o)
{
  const int head = blockIdx.x;
  const int tid  = threadIdx.x;
  __shared__ float sr[64], sk[64], sw[64], su[64];
  su[tid] = u[head * 64 + tid];
  float hs[64];
  #pragma unroll
  for (int i = 0; i < 64; i++) hs[i] = 0.0f;
  __syncthreads();
  for (int t = 0; t < T_SEQ; t++) {
    const int base = t * H_DIM + head * 64;
    sr[tid] = r[base + tid];
    sk[tid] = k[base + tid];
    sw[tid] = ew[base + tid];
    __syncthreads();
    const float vv = v[base + tid];
    float acc = 0.0f;
    #pragma unroll
    for (int j = 0; j < 64; j++) {
      const float kv = sk[j] * vv;
      acc  += sr[j] * (hs[j] + su[j] * kv);
      hs[j] = hs[j] * sw[j] + kv;
    }
    o[base + tid] = acc;
    __syncthreads();
  }
}

// GroupNorm over each head's 64 channels, then * g, -> bf16
__global__ void gn_kernel(const float* __restrict__ o,
                          const float* __restrict__ g,
                          const float* __restrict__ gw,
                          const float* __restrict__ gb,
                          bf16* __restrict__ og, int total)
{
  const int idx = blockIdx.x * blockDim.x + threadIdx.x;
  if (idx >= total) return;
  const int t = idx / NH_, hh = idx % NH_;
  const float* p = o + (size_t)t * H_DIM + hh * D_;
  float m = 0.0f;
  for (int d = 0; d < D_; d++) m += p[d];
  m *= (1.0f / D_);
  float var = 0.0f;
  for (int d = 0; d < D_; d++) { const float z = p[d] - m; var += z * z; }
  var *= (1.0f / D_);
  const float inv = rsqrtf(var + 1e-5f);
  for (int d = 0; d < D_; d++) {
    const int c = hh * D_ + d;
    const float y = (p[d] - m) * inv * gw[c] + gb[c];
    og[(size_t)t * H_DIM + c] = tobf(y * g[(size_t)t * H_DIM + c]);
  }
}

// ---------------------------------------------------------------------------
extern "C" void kernel_launch(void* const* d_in, const int* in_sizes, int n_in,
                              void* d_out, int out_size, void* d_ws, size_t ws_size,
                              hipStream_t stream)
{
  (void)in_sizes; (void)n_in; (void)out_size; (void)ws_size;
  const float* x        = (const float*)d_in[0];
  const float* ln1_w    = (const float*)d_in[1];
  const float* maa_x    = (const float*)d_in[2];
  const float* maa_w    = (const float*)d_in[3];
  const float* maa_k    = (const float*)d_in[4];
  const float* maa_v    = (const float*)d_in[5];
  const float* maa_r    = (const float*)d_in[6];
  const float* maa_g    = (const float*)d_in[7];
  const float* tm_w1    = (const float*)d_in[8];
  const float* tm_w2    = (const float*)d_in[9];
  const float* time_dec = (const float*)d_in[10];
  const float* td_w1    = (const float*)d_in[11];
  const float* td_w2    = (const float*)d_in[12];
  const float* Wr       = (const float*)d_in[13];
  const float* Wk       = (const float*)d_in[14];
  const float* Wv       = (const float*)d_in[15];
  const float* Wg       = (const float*)d_in[16];
  const float* bonus    = (const float*)d_in[17];
  const float* gn_w     = (const float*)d_in[18];
  const float* gn_b     = (const float*)d_in[19];
  const float* Wo       = (const float*)d_in[20];
  const float* ln2_w    = (const float*)d_in[21];
  const float* maa_kf   = (const float*)d_in[22];
  const float* maa_rf   = (const float*)d_in[23];
  const float* Wfk      = (const float*)d_in[24];
  const float* Wfr      = (const float*)d_in[25];
  const float* Wfv      = (const float*)d_in[26];
  float* out = (float*)d_out;

  char* base = (char*)d_ws;
  size_t off = 0;
  auto alloc = [&](size_t bytes) -> void* {
    void* p = base + off;
    off += (bytes + 255) & ~(size_t)255;
    return p;
  };
  const size_t TH = (size_t)T_SEQ * H_DIM;

  // bf16 transposed weights
  bf16* wt_tm1 = (bf16*)alloc((size_t)160 * 1024 * 2);
  bf16* wt_tm2 = (bf16*)alloc((size_t)5 * 1024 * 32 * 2);
  bf16* wt_td1 = (bf16*)alloc((size_t)64 * 1024 * 2);
  bf16* wt_td2 = (bf16*)alloc((size_t)1024 * 64 * 2);
  bf16* wt_r   = (bf16*)alloc((size_t)H_DIM * H_DIM * 2);
  bf16* wt_k   = (bf16*)alloc((size_t)H_DIM * H_DIM * 2);
  bf16* wt_v   = (bf16*)alloc((size_t)H_DIM * H_DIM * 2);
  bf16* wt_g   = (bf16*)alloc((size_t)H_DIM * H_DIM * 2);
  bf16* wt_o   = (bf16*)alloc((size_t)H_DIM * H_DIM * 2);
  bf16* wt_fr  = (bf16*)alloc((size_t)H_DIM * H_DIM * 2);
  bf16* wt_fk  = (bf16*)alloc((size_t)H_DIM * FF_ * 2);
  bf16* wt_fv  = (bf16*)alloc((size_t)FF_ * H_DIM * 2);
  // activations
  float* h1    = (float*)alloc(TH * 4);
  float* delta = (float*)alloc(TH * 4);
  bf16*  xxx   = (bf16*)alloc(TH * 2);
  bf16*  mixt  = (bf16*)alloc((size_t)T_SEQ * 160 * 2);
  bf16*  xw    = (bf16*)alloc(TH * 2);
  bf16*  xk    = (bf16*)alloc(TH * 2);
  bf16*  xv    = (bf16*)alloc(TH * 2);
  bf16*  xr    = (bf16*)alloc(TH * 2);
  bf16*  xg    = (bf16*)alloc(TH * 2);
  float* rr    = (float*)alloc(TH * 4);
  float* kk    = (float*)alloc(TH * 4);
  float* vv    = (float*)alloc(TH * 4);
  float* gg    = (float*)alloc(TH * 4);
  float* ew    = (float*)alloc(TH * 4);
  bf16*  wtan  = (bf16*)alloc((size_t)T_SEQ * 64 * 2);
  float* o_    = (float*)alloc(TH * 4);
  bf16*  og    = (bf16*)alloc(TH * 2);
  float* x2    = (float*)alloc(TH * 4);
  float* h2    = (float*)alloc(TH * 4);
  bf16*  xk2   = (bf16*)alloc(TH * 2);
  bf16*  xr2   = (bf16*)alloc(TH * 2);
  bf16*  kkb   = (bf16*)alloc((size_t)T_SEQ * FF_ * 2);
  float* sig   = (float*)alloc(TH * 4);

  auto conv = [&](const float* W, bf16* Wt, int K, int N) {
    const int n = K * N;
    wconv_kernel<<<(n + 255) / 256, 256, 0, stream>>>(W, Wt, K, N);
  };

#define GEMM(EPI, A_, LDA_, BT_, N_, K_, OF_, OB_, LDC_, P0_, P1_, P2_)        \
  gemm_bf16_kernel<EPI><<<dim3(T_SEQ / 64, ((N_) + 63) / 64), 128, 0, stream>>>( \
      A_, LDA_, BT_, T_SEQ, N_, K_, OF_, OB_, LDC_, P0_, P1_, P2_)

  const float* FNUL = nullptr;
  float* FONUL = nullptr;
  bf16*  BONUL = nullptr;

  // ---- weight conversion (+transpose to [N][K]) ----
  conv(tm_w1, wt_tm1, 1024, 160);
  for (int f = 0; f < 5; f++)
    conv(tm_w2 + (size_t)f * 32 * 1024, wt_tm2 + (size_t)f * 1024 * 32, 32, 1024);
  conv(td_w1, wt_td1, 1024, 64);
  conv(td_w2, wt_td2, 64, 1024);
  conv(Wr, wt_r, 1024, 1024);
  conv(Wk, wt_k, 1024, 1024);
  conv(Wv, wt_v, 1024, 1024);
  conv(Wg, wt_g, 1024, 1024);
  conv(Wo, wt_o, 1024, 1024);
  conv(Wfr, wt_fr, 1024, 1024);
  conv(Wfk, wt_fk, 1024, 4096);
  conv(Wfv, wt_fv, 4096, 1024);

  // ---- time mixing ----
  rmsnorm_kernel<<<T_SEQ, 256, 0, stream>>>(x, ln1_w, h1);
  shift1_kernel<<<(int)((TH + 255) / 256), 256, 0, stream>>>(h1, maa_x, delta, xxx, (int)TH);

  // mix = tanh(xxx @ tm_w1)  -> bf16 [T,160]
  GEMM(E_TANH_BF, xxx, H_DIM, wt_tm1, 160, H_DIM, FONUL, mixt, 160, FNUL, FNUL, FNUL);

  // x_f = h + delta*(maa_f + mix_f @ tm_w2[f])  -> bf16 [T,H]
  bf16* xouts[5]       = { xw, xk, xv, xr, xg };
  const float* maas[5] = { maa_w, maa_k, maa_v, maa_r, maa_g };
  for (int f = 0; f < 5; f++) {
    GEMM(E_MIX, mixt + f * 32, 160, wt_tm2 + (size_t)f * 1024 * 32, H_DIM, 32,
         FONUL, xouts[f], H_DIM, h1, delta, maas[f]);
  }

  GEMM(E_SIGSCALE, xr, H_DIM, wt_r, H_DIM, H_DIM, rr, BONUL, H_DIM, FNUL, FNUL, FNUL);
  GEMM(E_NONE,     xk, H_DIM, wt_k, H_DIM, H_DIM, kk, BONUL, H_DIM, FNUL, FNUL, FNUL);
  GEMM(E_NONE,     xv, H_DIM, wt_v, H_DIM, H_DIM, vv, BONUL, H_DIM, FNUL, FNUL, FNUL);
  GEMM(E_SILU,     xg, H_DIM, wt_g, H_DIM, H_DIM, gg, BONUL, H_DIM, FNUL, FNUL, FNUL);
  GEMM(E_TANH_BF,  xw, H_DIM, wt_td1, 64, H_DIM, FONUL, wtan, 64, FNUL, FNUL, FNUL);
  GEMM(E_DECAY,    wtan, 64, wt_td2, H_DIM, 64, ew, BONUL, H_DIM, time_dec, FNUL, FNUL);

  scan_kernel<<<NH_, 64, 0, stream>>>(rr, kk, vv, ew, bonus, o_);

  gn_kernel<<<(T_SEQ * NH_ + 255) / 256, 256, 0, stream>>>(o_, gg, gn_w, gn_b, og,
                                                           T_SEQ * NH_);

  // x2 = x + (gn(o)*g) @ Wo
  GEMM(E_ADDRES, og, H_DIM, wt_o, H_DIM, H_DIM, x2, BONUL, H_DIM, x, FNUL, FNUL);

  // ---- channel mixing ----
  rmsnorm_kernel<<<T_SEQ, 256, 0, stream>>>(x2, ln2_w, h2);
  shift2_kernel<<<(int)((TH + 255) / 256), 256, 0, stream>>>(h2, maa_kf, maa_rf,
                                                             xk2, xr2, (int)TH);

  GEMM(E_RELU2_BF, xk2, H_DIM, wt_fk, FF_, H_DIM, FONUL, kkb, FF_, FNUL, FNUL, FNUL);
  GEMM(E_SIGMOID,  xr2, H_DIM, wt_fr, H_DIM, H_DIM, sig, BONUL, H_DIM, FNUL, FNUL, FNUL);
  // out = x2 + sig * (kk @ Wfv)
  GEMM(E_FFNOUT,   kkb, FF_, wt_fv, H_DIM, FF_, out, BONUL, H_DIM, x2, sig, FNUL);

#undef GEMM
}
